// Int4MoePluginModule_69715909149344
// MI455X (gfx1250) — matrix operand order repaired and checked by reading, not compiled, not run on hardware
//
#include <hip/hip_runtime.h>
#include <hip/hip_bf16.h>
#include <math.h>

typedef _Float16 v16h __attribute__((ext_vector_type(16)));
typedef float    v8f  __attribute__((ext_vector_type(8)));

#define NEXP 8
#define HDIM 2048
#define IDIM 1408
#define GSZ  128
#define ZP   8

// ---------------------------------------------------------------------------
// WMMA fragment loaders from LDS (wave32, 16x16x32 f16).
// A-matrix 16x32 per-lane layout (ISA 7.12.2):
//   lanes 0-15: M=lane,    K in {0..7, 16..23}
//   lanes 16-31: M=lane-16, K in {8..15, 24..31}
// ---------------------------------------------------------------------------
__device__ __forceinline__ v16h ld_fragA(const _Float16* base, int stride) {
    int lane  = threadIdx.x & 31;
    int row   = lane & 15;
    int hlf   = lane >> 4;
    const _Float16* r = base + row * stride;
    union { v16h h; unsigned int u[8]; } f;
#pragma unroll
    for (int p = 0; p < 8; ++p) {
        int k = ((p & 4) << 2) + hlf * 8 + ((p & 3) << 1);
        f.u[p] = *(const unsigned int*)(r + k);
    }
    return f.h;
}

// B-matrix 32x16: lanes 0-15 hold K in {0..7, 16..23} (N=lane), lanes 16-31 K in {8..15, 24..31}
__device__ __forceinline__ v16h ld_fragB(const _Float16* base, int stride) {
    int lane = threadIdx.x & 31;
    int row  = lane & 15;
    int hlf  = lane >> 4;
    const _Float16* r = base + row * stride;
    union { v16h h; unsigned int u[8]; } f;
#pragma unroll
    for (int p = 0; p < 8; ++p)
        f.u[p] = *(const unsigned int*)(r + ((p & 4) << 2) + hlf * 8 + ((p & 3) << 1));
    return f.h;
}

// Dequant 4 int4-in-int32 with one scale -> 4 packed f16.
// (float)q*s + (-8s) lowers to v_cvt_f32_i32 + v_fma_mixlo/hi_f16.
__device__ __forceinline__ uint2 dq4(int4 q, float s) {
    float ns = -8.f * s;
    union { _Float16 h[4]; uint2 u; } p;
    p.h[0] = (_Float16)((float)q.x * s + ns);
    p.h[1] = (_Float16)((float)q.y * s + ns);
    p.h[2] = (_Float16)((float)q.z * s + ns);
    p.h[3] = (_Float16)((float)q.w * s + ns);
    return p.u;
}

// ---------------------------------------------------------------------------
// Routing: top-2 + softmax per token
// ---------------------------------------------------------------------------
__global__ void k_route(const float* __restrict__ logits, int T,
                        int* __restrict__ counts,
                        int* __restrict__ tok_e, float* __restrict__ tok_w) {
    int t = blockIdx.x * blockDim.x + threadIdx.x;
    if (t >= T) return;
    const float* l = logits + (size_t)t * NEXP;
    float v0 = -1e30f; int e0 = 0;
#pragma unroll
    for (int e = 0; e < NEXP; ++e) { float v = l[e]; if (v > v0) { v0 = v; e0 = e; } }
    float v1 = -1e30f; int e1 = 0;
#pragma unroll
    for (int e = 0; e < NEXP; ++e) { float v = l[e]; if (e != e0 && v > v1) { v1 = v; e1 = e; } }
    float w0 = 1.f / (1.f + __expf(v1 - v0));
    tok_e[2 * t] = e0; tok_e[2 * t + 1] = e1;
    tok_w[2 * t] = w0; tok_w[2 * t + 1] = 1.f - w0;
    atomicAdd(&counts[e0], 1);
    atomicAdd(&counts[e1], 1);
}

__global__ void k_scan(const int* __restrict__ counts,
                       int* __restrict__ offs, int* __restrict__ fill) {
    if (threadIdx.x == 0) {
        int acc = 0;
        for (int e = 0; e < NEXP; ++e) { offs[e] = acc; acc += counts[e]; }
    }
    if (threadIdx.x < NEXP) fill[threadIdx.x] = 0;
}

__global__ void k_scatter(int T, const int* __restrict__ tok_e,
                          const float* __restrict__ tok_w,
                          const int* __restrict__ offs, int* __restrict__ fill,
                          int* __restrict__ slot_token, float* __restrict__ slot_w) {
    int t = blockIdx.x * blockDim.x + threadIdx.x;
    if (t >= T) return;
#pragma unroll
    for (int j = 0; j < 2; ++j) {
        int e = tok_e[2 * t + j];
        int p = atomicAdd(&fill[e], 1);
        int s = offs[e] + p;
        slot_token[s] = t;
        slot_w[s]     = tok_w[2 * t + j];
    }
}

// ---------------------------------------------------------------------------
// Gate/Up GEMM + SwiGLU.  Block: 32 slots x 64 intermediate cols, 8 waves.
// Wave w: m-tile = w>>2, n-tile = w&3; accumulates gate AND up 16x16 tiles.
// Double-buffered LDS, ONE barrier per K-step; global loads for step i+1
// issued before the barrier so they overlap fragment loads + WMMA of step i.
// Quant-group outer loop (4 K-steps/group) hoists scale loads; inner loop is
// fully unrolled so LDS buffer parity (0,1,0,1) is compile-time.
// ---------------------------------------------------------------------------
struct GUr { float4 a; int4 qg0, qu0, qg1, qu1; };

__global__ __launch_bounds__(256) void k_gateup(
    const float* __restrict__ x,          // [T, H]
    const int*   __restrict__ guq,        // [E, 2I, H] int4-in-int32
    const float* __restrict__ gus,        // [E, 2I, H/GSZ]
    const int*   __restrict__ counts, const int* __restrict__ offs,
    const int*   __restrict__ slot_token,
    _Float16*    __restrict__ act)        // [2T, I]
{
    constexpr int BM = 32, BN = 64, BK = 32, LDST = 36;
    constexpr int NIT = HDIM / BK, NGRP = HDIM / GSZ, SPG = GSZ / BK; // 64,16,4
    constexpr int ASZ = BM * LDST, WSZ = BN * LDST;
    __shared__ __attribute__((aligned(16))) _Float16 As[2 * ASZ];
    __shared__ __attribute__((aligned(16))) _Float16 Wg[2 * WSZ];
    __shared__ __attribute__((aligned(16))) _Float16 Wu[2 * WSZ];

    int e   = blockIdx.z;
    int cnt = counts[e];
    int m0  = blockIdx.x * BM;
    if (m0 >= cnt) return;
    int base = offs[e];
    int n0   = blockIdx.y * BN;

    int tid = threadIdx.x, wave = tid >> 5, lane = tid & 31;
    int mt = wave >> 2, nt = wave & 3;

    const int*   qb = guq + (size_t)e * (2 * IDIM) * HDIM;
    const float* sb = gus + (size_t)e * (2 * IDIM) * (HDIM / GSZ);

    // A-staging map: thread -> (row am, 4 consecutive k at ak)
    int  am = tid >> 3;
    int  ak = (tid & 7) * 4;
    bool avalid = (m0 + am) < cnt;
    int  aslot  = avalid ? (base + m0 + am) : base;
    const float* xrow = x + (size_t)slot_token[aslot] * HDIM + ak;
    float amask = avalid ? 1.f : 0.f;
    _Float16* adst = As + am * LDST + ak;

    // W-staging map: 2 tasks/thread over 64 rows x 8 k-quads
    const int*   qgp[2]; const int* qup[2];
    const float* sgp[2]; const float* sup[2];
    _Float16* wgd[2]; _Float16* wud[2];
#pragma unroll
    for (int j = 0; j < 2; ++j) {
        int task = tid + j * 256;
        int n = task >> 3, kq = (task & 7) * 4;
        qgp[j] = qb + (size_t)(n0 + n) * HDIM + kq;
        qup[j] = qb + (size_t)(IDIM + n0 + n) * HDIM + kq;
        sgp[j] = sb + (size_t)(n0 + n) * (HDIM / GSZ);
        sup[j] = sb + (size_t)(IDIM + n0 + n) * (HDIM / GSZ);
        wgd[j] = Wg + n * LDST + kq;
        wud[j] = Wu + n * LDST + kq;
    }

    v8f cg = {}; v8f cu = {};

    auto loadG = [&](int it, GUr& r) {
        int k0 = it * BK;
        r.a   = *(const float4*)(xrow + k0);
        r.qg0 = *(const int4*)(qgp[0] + k0);
        r.qu0 = *(const int4*)(qup[0] + k0);
        r.qg1 = *(const int4*)(qgp[1] + k0);
        r.qu1 = *(const int4*)(qup[1] + k0);
    };
    auto stage = [&](int buf, const GUr& r,
                     float s0g, float s0u, float s1g, float s1u) {
        union { _Float16 h[4]; uint2 u; } ap;
        ap.h[0] = (_Float16)(r.a.x * amask);
        ap.h[1] = (_Float16)(r.a.y * amask);
        ap.h[2] = (_Float16)(r.a.z * amask);
        ap.h[3] = (_Float16)(r.a.w * amask);
        *(uint2*)(adst + buf * ASZ)   = ap.u;
        *(uint2*)(wgd[0] + buf * WSZ) = dq4(r.qg0, s0g);
        *(uint2*)(wud[0] + buf * WSZ) = dq4(r.qu0, s0u);
        *(uint2*)(wgd[1] + buf * WSZ) = dq4(r.qg1, s1g);
        *(uint2*)(wud[1] + buf * WSZ) = dq4(r.qu1, s1u);
    };
    auto mm = [&](int buf) {
        v16h a  = ld_fragA(As + buf * ASZ + mt * 16 * LDST, LDST);
        v16h bg = ld_fragB(Wg + buf * WSZ + nt * 16 * LDST, LDST);
        v16h bu = ld_fragB(Wu + buf * WSZ + nt * 16 * LDST, LDST);
        cg = __builtin_amdgcn_wmma_f32_16x16x32_f16(false, a, false, bg, (short)0, cg, false, false);
        cu = __builtin_amdgcn_wmma_f32_16x16x32_f16(false, a, false, bu, (short)0, cu, false, false);
    };

    GUr r;
    loadG(0, r);
    for (int g = 0; g < NGRP; ++g) {
        float s0g = sgp[0][g], s0u = sup[0][g];
        float s1g = sgp[1][g], s1u = sup[1][g];
#pragma unroll
        for (int s = 0; s < SPG; ++s) {
            int it  = g * SPG + s;
            int buf = s & 1;
            stage(buf, r, s0g, s0u, s1g, s1u);
            if (it + 1 < NIT) loadG(it + 1, r);
            __syncthreads();
            mm(buf);
        }
    }

    // SwiGLU epilogue (C layout: elem r -> M = r + (lane>>4)*8, N = lane&15)
    int nCol = n0 + nt * 16 + (lane & 15);
    int hlf  = lane >> 4;
#pragma unroll
    for (int rr = 0; rr < 8; ++rr) {
        int m = mt * 16 + rr + hlf * 8;
        if (m0 + m < cnt) {
            float gv = cg[rr], uv = cu[rr];
            float si = gv * __builtin_amdgcn_rcpf(1.f + __expf(-gv));
            act[(size_t)(base + m0 + m) * IDIM + nCol] = (_Float16)(si * uv);
        }
    }
}

// ---------------------------------------------------------------------------
// Down GEMM.  Block: 32 slots x 128 H-cols, 8 waves; wave w owns n-tile w and
// both m-tiles.  Same double-buffered single-barrier pipeline with per-group
// scale hoisting.  Epilogue scales by routing weight, atomicAdd into out
// (exactly 2 commutative f32 adds per element -> deterministic).
// ---------------------------------------------------------------------------
struct DNr { uint2 a; int4 q0, q1, q2, q3; };

__global__ __launch_bounds__(256) void k_down(
    const int*   __restrict__ dq,         // [E, H, I]
    const float* __restrict__ dsc,        // [E, H, I/GSZ]
    const int*   __restrict__ counts, const int* __restrict__ offs,
    const int*   __restrict__ slot_token, const float* __restrict__ slot_w,
    const _Float16* __restrict__ act,     // [2T, I]
    float* __restrict__ out)              // [T, H]
{
    constexpr int BM = 32, BN = 128, BK = 32, LDST = 36;
    constexpr int NIT = IDIM / BK, NGRP = IDIM / GSZ, SPG = GSZ / BK; // 44,11,4
    constexpr int ASZ = BM * LDST, WSZ = BN * LDST;
    __shared__ __attribute__((aligned(16))) _Float16 As[2 * ASZ];
    __shared__ __attribute__((aligned(16))) _Float16 Wd[2 * WSZ];

    int e   = blockIdx.z;
    int cnt = counts[e];
    int m0  = blockIdx.x * BM;
    if (m0 >= cnt) return;
    int base = offs[e];
    int n0   = blockIdx.y * BN;

    int tid = threadIdx.x, wave = tid >> 5, lane = tid & 31;

    const int*   qb = dq  + (size_t)e * HDIM * IDIM;
    const float* sb = dsc + (size_t)e * HDIM * (IDIM / GSZ);

    // A-staging map (act rows, f16): thread -> (row, 4 halves)
    int  am = tid >> 3;
    int  ak = (tid & 7) * 4;
    bool avalid = (m0 + am) < cnt;
    int  aslot  = avalid ? (base + m0 + am) : base;
    const _Float16* arow = act + (size_t)aslot * IDIM + ak;
    unsigned amsk = avalid ? 0xFFFFFFFFu : 0u;
    _Float16* adst = As + am * LDST + ak;

    // W-staging map: 4 tasks/thread over 128 rows x 8 k-quads
    const int*   qp[4];
    const float* sp[4];
    _Float16* wdd[4];
#pragma unroll
    for (int j = 0; j < 4; ++j) {
        int task = tid + j * 256;
        int n = task >> 3, kq = (task & 7) * 4;
        qp[j]  = qb + (size_t)(n0 + n) * IDIM + kq;
        sp[j]  = sb + (size_t)(n0 + n) * (IDIM / GSZ);
        wdd[j] = Wd + n * LDST + kq;
    }

    v8f c0 = {}; v8f c1 = {};

    auto loadG = [&](int it, DNr& r) {
        int k0 = it * BK;
        r.a  = *(const uint2*)(arow + k0);
        r.q0 = *(const int4*)(qp[0] + k0);
        r.q1 = *(const int4*)(qp[1] + k0);
        r.q2 = *(const int4*)(qp[2] + k0);
        r.q3 = *(const int4*)(qp[3] + k0);
    };
    auto stage = [&](int buf, const DNr& r, const float* sv) {
        uint2 av; av.x = r.a.x & amsk; av.y = r.a.y & amsk;
        *(uint2*)(adst + buf * ASZ)   = av;
        *(uint2*)(wdd[0] + buf * WSZ) = dq4(r.q0, sv[0]);
        *(uint2*)(wdd[1] + buf * WSZ) = dq4(r.q1, sv[1]);
        *(uint2*)(wdd[2] + buf * WSZ) = dq4(r.q2, sv[2]);
        *(uint2*)(wdd[3] + buf * WSZ) = dq4(r.q3, sv[3]);
    };
    auto mm = [&](int buf) {
        v16h a0 = ld_fragA(As + buf * ASZ, LDST);
        v16h a1 = ld_fragA(As + buf * ASZ + 16 * LDST, LDST);
        v16h b  = ld_fragB(Wd + buf * WSZ + wave * 16 * LDST, LDST);
        c0 = __builtin_amdgcn_wmma_f32_16x16x32_f16(false, a0, false, b, (short)0, c0, false, false);
        c1 = __builtin_amdgcn_wmma_f32_16x16x32_f16(false, a1, false, b, (short)0, c1, false, false);
    };

    DNr r;
    loadG(0, r);
    for (int g = 0; g < NGRP; ++g) {
        float sv[4];
#pragma unroll
        for (int j = 0; j < 4; ++j) sv[j] = sp[j][g];
#pragma unroll
        for (int s = 0; s < SPG; ++s) {
            int it  = g * SPG + s;
            int buf = s & 1;
            stage(buf, r, sv);
            if (it + 1 < NIT) loadG(it + 1, r);
            __syncthreads();
            mm(buf);
        }
    }

    int nCol = n0 + wave * 16 + (lane & 15);
    int hlf  = lane >> 4;
#pragma unroll
    for (int rr = 0; rr < 8; ++rr) {
        int mA = rr + hlf * 8;
        int mB = 16 + mA;
        if (m0 + mA < cnt) {
            int s = base + m0 + mA;
            atomicAdd(&out[(size_t)slot_token[s] * HDIM + nCol], c0[rr] * slot_w[s]);
        }
        if (m0 + mB < cnt) {
            int s = base + m0 + mB;
            atomicAdd(&out[(size_t)slot_token[s] * HDIM + nCol], c1[rr] * slot_w[s]);
        }
    }
}

// ---------------------------------------------------------------------------
extern "C" void kernel_launch(void* const* d_in, const int* in_sizes, int n_in,
                              void* d_out, int out_size, void* d_ws, size_t ws_size,
                              hipStream_t stream) {
    const float* logits = (const float*)d_in[0];   // [T, E]
    const float* x      = (const float*)d_in[1];   // [T, H]
    const int*   guq    = (const int*)  d_in[2];   // [E, 2I, H]
    const float* gus    = (const float*)d_in[3];   // [E, 2I, H/GSZ]
    const int*   dqw    = (const int*)  d_in[4];   // [E, H, I]
    const float* dsc    = (const float*)d_in[5];   // [E, H, I/GSZ]
    float* out = (float*)d_out;

    const int T = in_sizes[0] / NEXP;              // 2048
    const int S = 2 * T;                           // total (token, expert) slots

    // Workspace layout
    char* ws = (char*)d_ws;
    int*   counts     = (int*)ws;                    ws += 16 * sizeof(int);
    int*   offs       = (int*)ws;                    ws += 16 * sizeof(int);
    int*   fill       = (int*)ws;                    ws += 16 * sizeof(int);
    int*   tok_e      = (int*)ws;                    ws += (size_t)S * sizeof(int);
    float* tok_w      = (float*)ws;                  ws += (size_t)S * sizeof(float);
    int*   slot_token = (int*)ws;                    ws += (size_t)S * sizeof(int);
    float* slot_w     = (float*)ws;                  ws += (size_t)S * sizeof(float);
    size_t off = (size_t)(ws - (char*)d_ws);
    off = (off + 15) & ~(size_t)15;
    _Float16* act = (_Float16*)((char*)d_ws + off); // [S, I] f16

    hipMemsetAsync(out, 0, (size_t)out_size * sizeof(float), stream);
    hipMemsetAsync(counts, 0, NEXP * sizeof(int), stream);

    int tb = (T + 255) / 256;
    k_route  <<<tb, 256, 0, stream>>>(logits, T, counts, tok_e, tok_w);
    k_scan   <<<1, 32, 0, stream>>>(counts, offs, fill);
    k_scatter<<<tb, 256, 0, stream>>>(T, tok_e, tok_w, offs, fill, slot_token, slot_w);

    dim3 gGU((T + 31) / 32, IDIM / 64, NEXP);      // 64 x 22 x 8
    k_gateup<<<gGU, 256, 0, stream>>>(x, guq, gus, counts, offs, slot_token, act);

    dim3 gDN((T + 31) / 32, HDIM / 128, NEXP);     // 64 x 16 x 8
    k_down<<<gDN, 256, 0, stream>>>(dqw, dsc, counts, offs, slot_token, slot_w, act, out);
}
